// TransformerLNN_15487652069431
// MI455X (gfx1250) — matrix operand order repaired and checked by reading, not compiled, not run on hardware
//
#include <hip/hip_runtime.h>
#include <hip/hip_bf16.h>
#include <math.h>

// ---------------------------------------------------------------------------
// MI455X / gfx1250, wave32. Heavy matmuls -> v_wmma_f32_16x16x32_bf16.
// bf16 activations + pre-transposed bf16 weights in global memory so LDS
// staging is pure byte copies via GLOBAL_LOAD_ASYNC_TO_LDS_B128 (ASYNCcnt).
// f32 kept for residual / LayerNorm / scan paths.
// ---------------------------------------------------------------------------

typedef __bf16 v16bf  __attribute__((ext_vector_type(16)));
typedef __bf16 bf16x8 __attribute__((ext_vector_type(8)));
typedef float  v8f    __attribute__((ext_vector_type(8)));
typedef int    v4i    __attribute__((ext_vector_type(4)));

#define NH    8
#define DH    128
#define LSEQ  2048
#define HID   1024
#define MROWS 4096      // B*L

__device__ __forceinline__ unsigned short f2bf(float f) {
    unsigned int u = __float_as_uint(f);
    u += 0x7FFFu + ((u >> 16) & 1u);      // round-to-nearest-even
    return (unsigned short)(u >> 16);
}
__device__ __forceinline__ float sigf(float x) { return 1.0f / (1.0f + __expf(-x)); }

// ------------------------- async-to-LDS (guarded) --------------------------
#if defined(__gfx1250__) && __has_builtin(__builtin_amdgcn_global_load_async_to_lds_b128)
#define HAVE_ASYNC_LDS 1
#else
#define HAVE_ASYNC_LDS 0
#endif

#if HAVE_ASYNC_LDS
typedef __attribute__((address_space(1))) v4i* g4p;   // global int4*
typedef __attribute__((address_space(3))) v4i* l4p;   // LDS int4*
__device__ __forceinline__ void async_b128(void* lds, const void* g) {
    // generic LDS pointer -> 32-bit LDS offset -> AS(3) pointer
    l4p l = (l4p)(unsigned int)(unsigned long long)lds;
    // generic global pointer -> AS(1) pointer (drop const to match builtin)
    g4p gp = (g4p)(unsigned long long)g;
    __builtin_amdgcn_global_load_async_to_lds_b128(gp, l, 0, 0);
}
#endif
__device__ __forceinline__ void wait_async_lds() {
#if HAVE_ASYNC_LDS
#if __has_builtin(__builtin_amdgcn_s_wait_asynccnt)
    __builtin_amdgcn_s_wait_asynccnt(0);
#else
    asm volatile("s_wait_asynccnt 0x0" ::: "memory");
#endif
#endif
}
// copy 16 bf16 (32B) global -> LDS
__device__ __forceinline__ void copy32B(unsigned short* lds, const unsigned short* g) {
#if HAVE_ASYNC_LDS
    async_b128(lds, g);
    async_b128(lds + 8, g + 8);
#else
    *(uint4*)lds       = *(const uint4*)g;
    *(uint4*)(lds + 8) = *(const uint4*)(g + 8);
#endif
}

enum { ACT_NONE = 0, ACT_SILU = 1, ACT_TANH = 2 };

// ---------------------------------------------------------------------------
// GEMM: C[M,N] = act(A[M,K] @ B[K,N] + bias), A bf16 [M,K], Bt bf16 [N,K].
// 256 thr / 8 waves; 128x128 C tile, BK=32, double-buffered LDS.
// Wave grid 2(M)x4(N): 4 A-frags x 2 B-frags = 8 WMMA / wave / K-step.
// Outputs: Cf (f32, optional) and Cb (bf16, optional).
// Requires M%128==0, N%128==0, K%32==0.
// ---------------------------------------------------------------------------
template <int ACT>
__global__ __launch_bounds__(256)
void gemm_bf16t_kernel(const unsigned short* __restrict__ A,
                       const unsigned short* __restrict__ Bt,
                       const float* __restrict__ bias,
                       float* __restrict__ Cf, unsigned short* __restrict__ Cb,
                       int M, int N, int K) {
    __shared__ __align__(32) unsigned short As[2][128][32];
    __shared__ __align__(32) unsigned short Bs[2][128][32];   // [n][k]

    const int tid  = threadIdx.x;
    const int lane = tid & 31, wid = tid >> 5;
    const int waveM = wid >> 2, waveN = wid & 3;
    const int blkN = blockIdx.x, blkM = blockIdx.y;
    const int m = lane & 15, hb = lane >> 4;

    v8f acc[4][2];
#pragma unroll
    for (int f = 0; f < 4; ++f)
#pragma unroll
        for (int t = 0; t < 2; ++t) acc[f][t] = (v8f){0,0,0,0,0,0,0,0};

    // staging: each thread copies 32B of A tile and 32B of B tile per step
    const int srow = tid >> 1, soff = (tid & 1) * 16;
    const unsigned short* Ag = A  + (size_t)(blkM * 128 + srow) * K + soff;
    const unsigned short* Bg = Bt + (size_t)(blkN * 128 + srow) * K + soff;

    auto stage = [&](int buf, int kt) {
        const unsigned short* ga = Ag + (size_t)kt * 32;
        const unsigned short* gb = Bg + (size_t)kt * 32;
        copy32B(&As[buf][srow][soff], ga);
        copy32B(&Bs[buf][srow][soff], gb);
#if !HAVE_ASYNC_LDS
        __builtin_prefetch(ga + 32, 0, 3);
        __builtin_prefetch(gb + 32, 0, 3);
#endif
    };

    auto compute = [&](int buf) {
        v16bf af[4], bfr[2];
#pragma unroll
        for (int f = 0; f < 4; ++f) {
            const unsigned short* ap = &As[buf][waveM * 64 + f * 16 + m][hb * 16];
            bf16x8 lo = *(const bf16x8*)(ap);
            bf16x8 hi = *(const bf16x8*)(ap + 8);
            af[f] = __builtin_shufflevector(lo, hi, 0,1,2,3,4,5,6,7,8,9,10,11,12,13,14,15);
        }
#pragma unroll
        for (int t = 0; t < 2; ++t)
            bfr[t] = *(const v16bf*)&Bs[buf][waveN * 32 + t * 16 + m][hb * 16];
#pragma unroll
        for (int f = 0; f < 4; ++f)
#pragma unroll
            for (int t = 0; t < 2; ++t)
                acc[f][t] = __builtin_amdgcn_wmma_f32_16x16x32_bf16(
                    false, af[f], false, bfr[t], (short)0, acc[f][t], false, false);
    };

    const int nk = K >> 5;
    int buf = 0;
    stage(0, 0);
    for (int kt = 0; kt < nk; ++kt) {
        wait_async_lds();
        __syncthreads();
        if (kt + 1 < nk) stage(buf ^ 1, kt + 1);
        compute(buf);
        buf ^= 1;
    }

    // epilogue
#pragma unroll
    for (int f = 0; f < 4; ++f) {
        const int growb = blkM * 128 + waveM * 64 + f * 16 + hb * 8;
#pragma unroll
        for (int t = 0; t < 2; ++t) {
            const int gcol = blkN * 128 + waveN * 32 + t * 16 + m;
            const float bi = bias ? bias[gcol] : 0.0f;
#pragma unroll
            for (int r = 0; r < 8; ++r) {
                float v = acc[f][t][r] + bi;
                if (ACT == ACT_SILU) v *= sigf(v);
                if (ACT == ACT_TANH) v = tanhf(v);
                const size_t idx = (size_t)(growb + r) * N + gcol;
                if (Cf) Cf[idx] = v;
                if (Cb) Cb[idx] = f2bf(v);
            }
        }
    }
}

// ---------------------------------------------------------------------------
// Flash attention over bf16 qkv [B*L, 3H]. One block (128 thr / 4 waves) per
// (b, head, 64 q-rows); wave owns 16 q rows; both matmuls on WMMA bf16.
// Writes bf16 head-concat output [B*L, H].
// ---------------------------------------------------------------------------
__global__ __launch_bounds__(128)
void attn_kernel(const unsigned short* __restrict__ qkv, unsigned short* __restrict__ aoh) {
    __shared__ __align__(32) unsigned short Ks[32][DH];     // K rows, bf16
    __shared__ __align__(32) unsigned short Vt[DH][32];     // V transposed
    __shared__ __align__(32) unsigned short Pb[4][16][32];  // per-wave P tile

    const int tid  = threadIdx.x;
    const int lane = tid & 31, wid = tid >> 5;
    const int bid = blockIdx.x;         // B*NH*(L/64) = 512
    const int qt = bid & 31;
    const int h  = (bid >> 5) & 7;
    const int b  = bid >> 8;
    const int q0 = qt * 64;
    const int m  = lane & 15, hb = lane >> 4;
    const int kb2 = hb * 16;

    // preload Q A-fragments (bf16 direct)
    v16bf qf[4];
    {
        const unsigned short* qp =
            qkv + (size_t)(b * LSEQ + q0 + wid * 16 + m) * (3 * HID) + h * DH;
#pragma unroll
        for (int c = 0; c < 4; ++c) {
            const unsigned short* p = qp + c * 32 + hb * 16;
            bf16x8 lo = *(const bf16x8*)(p);
            bf16x8 hi = *(const bf16x8*)(p + 8);
            qf[c] = __builtin_shufflevector(lo, hi, 0,1,2,3,4,5,6,7,8,9,10,11,12,13,14,15);
        }
    }

    float mrow[8], lrow[8];
    v8f oacc[8];
#pragma unroll
    for (int r = 0; r < 8; ++r) { mrow[r] = -1e30f; lrow[r] = 0.0f; }
#pragma unroll
    for (int d = 0; d < 8; ++d) oacc[d] = (v8f){0,0,0,0,0,0,0,0};

    const int skk = tid >> 2, sc0 = (tid & 3) * 32;  // staging: 32 rows x 128 cols
    for (int kt = 0; kt < LSEQ / 32; ++kt) {
        const int kv0 = kt * 32;
        __syncthreads();
        {
            const unsigned short* kp =
                qkv + (size_t)(b * LSEQ + kv0 + skk) * (3 * HID) + HID + h * DH + sc0;
            // K tile: straight row copy (async-capable)
            copy32B(&Ks[skk][sc0], kp);
            copy32B(&Ks[skk][sc0 + 16], kp + 16);
            // V tile: transpose scatter (register path)
            const unsigned short* vp = kp + HID;
#pragma unroll
            for (int j = 0; j < 32; j += 8) {
                bf16x8 v = *(const bf16x8*)(vp + j);
#pragma unroll
                for (int e = 0; e < 8; ++e)
                    *(__bf16*)&Vt[sc0 + j + e][skk] = v[e];
            }
        }
        wait_async_lds();
        __syncthreads();

        // S = Q @ K^T: B[kk][n] = K[kv0+n][kk] = Ks[n][kk]
        v8f s0 = (v8f){0,0,0,0,0,0,0,0};
        v8f s1 = (v8f){0,0,0,0,0,0,0,0};
#pragma unroll
        for (int c = 0; c < 4; ++c) {
            v16bf bk0 = *(const v16bf*)&Ks[m][c * 32 + kb2];
            v16bf bk1 = *(const v16bf*)&Ks[16 + m][c * 32 + kb2];
            s0 = __builtin_amdgcn_wmma_f32_16x16x32_bf16(false, qf[c], false, bk0,
                                                         (short)0, s0, false, false);
            s1 = __builtin_amdgcn_wmma_f32_16x16x32_bf16(false, qf[c], false, bk1,
                                                         (short)0, s1, false, false);
        }

        // online softmax (rows per half-wave; reduce across 16 lanes)
        const float sc = 0.08838834764831845f;   // 1/sqrt(128)
#pragma unroll
        for (int r = 0; r < 8; ++r) {
            float a0 = s0[r] * sc, a1 = s1[r] * sc;
            float mx = fmaxf(a0, a1);
#pragma unroll
            for (int off = 1; off < 16; off <<= 1) mx = fmaxf(mx, __shfl_xor(mx, off, 16));
            float mnew  = fmaxf(mrow[r], mx);
            float scale = __expf(mrow[r] - mnew);
            float p0 = __expf(a0 - mnew), p1 = __expf(a1 - mnew);
            float ps = p0 + p1;
#pragma unroll
            for (int off = 1; off < 16; off <<= 1) ps += __shfl_xor(ps, off, 16);
            lrow[r] = lrow[r] * scale + ps;
            mrow[r] = mnew;
#pragma unroll
            for (int d = 0; d < 8; ++d) oacc[d][r] *= scale;
            Pb[wid][r + hb * 8][m]      = f2bf(p0);
            Pb[wid][r + hb * 8][m + 16] = f2bf(p1);
        }

        // P fragment (A layout) from per-wave LDS tile
        const unsigned short* pp = &Pb[wid][m][hb * 16];
        bf16x8 plo = *(const bf16x8*)(pp);
        bf16x8 phi = *(const bf16x8*)(pp + 8);
        v16bf pf = __builtin_shufflevector(plo, phi, 0,1,2,3,4,5,6,7,8,9,10,11,12,13,14,15);

        // O += P @ V : B[kk][n] = V[kv0+kk][d*16+n] = Vt[d*16+n][kk]
#pragma unroll
        for (int d = 0; d < 8; ++d) {
            v16bf bv = *(const v16bf*)&Vt[d * 16 + m][kb2];
            oacc[d] = __builtin_amdgcn_wmma_f32_16x16x32_bf16(false, pf, false, bv,
                                                              (short)0, oacc[d], false, false);
        }
    }

    // finalize: O / l, write bf16 [B*L, H], head h at cols h*DH
#pragma unroll
    for (int r = 0; r < 8; ++r) {
        float inv = 1.0f / lrow[r];
        const size_t row = (size_t)(b * LSEQ + q0 + wid * 16 + hb * 8 + r);
#pragma unroll
        for (int d = 0; d < 8; ++d)
            aoh[row * HID + h * DH + d * 16 + m] = f2bf(oacc[d][r] * inv);
    }
}

// ---------------------------------------------------------------------------
// Fused residual LayerNorm: out = LN(in0 + in1 [+ in2]) * g + be, f32 + bf16.
// ---------------------------------------------------------------------------
__global__ __launch_bounds__(256)
void ln_res_kernel(const float* __restrict__ in0, const float* __restrict__ in1,
                   const float* __restrict__ in2, const float* __restrict__ g,
                   const float* __restrict__ be, float* __restrict__ out,
                   unsigned short* __restrict__ outb) {
    __shared__ float red[256];
    const int row = blockIdx.x, tid = threadIdx.x;
    const size_t base = (size_t)row * HID;
    float v[4];
    float s = 0.0f;
#pragma unroll
    for (int j = 0; j < 4; ++j) {
        int i = tid + j * 256;
        float x = in0[base + i] + in1[base + i];
        if (in2) x += in2[base + i];
        v[j] = x; s += x;
    }
    red[tid] = s; __syncthreads();
    for (int o = 128; o > 0; o >>= 1) { if (tid < o) red[tid] += red[tid + o]; __syncthreads(); }
    const float mu = red[0] * (1.0f / HID);
    __syncthreads();
    s = 0.0f;
#pragma unroll
    for (int j = 0; j < 4; ++j) { float d = v[j] - mu; s += d * d; }
    red[tid] = s; __syncthreads();
    for (int o = 128; o > 0; o >>= 1) { if (tid < o) red[tid] += red[tid + o]; __syncthreads(); }
    const float rstd = rsqrtf(red[0] * (1.0f / HID) + 1e-5f);
#pragma unroll
    for (int j = 0; j < 4; ++j) {
        int i = tid + j * 256;
        float o = (v[j] - mu) * rstd * g[i] + be[i];
        out[base + i] = o;
        if (outb) outb[base + i] = f2bf(o);
    }
}

// alpha = sigmoid(h1 @ W_cg + b_cg): one wave per row (H->1 GEMV)
__global__ __launch_bounds__(256)
void cube_gate_gemv(const float* __restrict__ h1, const float* __restrict__ Wcg,
                    const float* __restrict__ bcg, float* __restrict__ alpha) {
    const int tid = threadIdx.x, lane = tid & 31, wid = tid >> 5;
    const int row = blockIdx.x * 8 + wid;
    const size_t base = (size_t)row * HID;
    float acc = 0.0f;
    for (int i = lane; i < HID; i += 32) acc += h1[base + i] * Wcg[i];
#pragma unroll
    for (int off = 16; off > 0; off >>= 1) acc += __shfl_xor(acc, off, 32);
    if (lane == 0) alpha[row] = sigf(acc + bcg[0]);
}

// y_cg = alpha*cand + (1-alpha)*h1
__global__ __launch_bounds__(256)
void cube_blend_kernel(const float* __restrict__ alpha, const float* __restrict__ cand,
                       const float* __restrict__ h1, float* __restrict__ ycg, int n) {
    int i = blockIdx.x * 256 + threadIdx.x;
    if (i < n) {
        float a = alpha[i >> 10];
        ycg[i] = a * cand[i] + (1.0f - a) * h1[i];
    }
}

// ---------------------------------------------------------------------------
// Liquid time-constant scan: strictly sequential over L; single workgroup.
// ---------------------------------------------------------------------------
__global__ __launch_bounds__(1024)
void ltc_scan_kernel(const float* __restrict__ ycg, const float* __restrict__ times,
                     const float* __restrict__ Wlb, const float* __restrict__ blb,
                     const float* __restrict__ Wlt, const float* __restrict__ blt,
                     const float* __restrict__ Wlg, const float* __restrict__ blg,
                     const float* __restrict__ Wlh, const float* __restrict__ blh,
                     float* __restrict__ yhid) {
    __shared__ float xs[2][HID];
    __shared__ float st[2][HID];
    __shared__ float ft[2][HID];
    const int j = threadIdx.x;
    st[0][j] = 0.0f; st[1][j] = 0.0f;
    __syncthreads();

    for (int t = 0; t < LSEQ; ++t) {
        xs[0][j] = ycg[(size_t)(0 * LSEQ + t) * HID + j];
        xs[1][j] = ycg[(size_t)(1 * LSEQ + t) * HID + j];
        __syncthreads();
#pragma unroll
        for (int b = 0; b < 2; ++b) {
            float acc = blb[j];
#pragma unroll 4
            for (int i = 0; i < HID; ++i) acc += xs[b][i] * Wlb[(size_t)i * HID + j];
#pragma unroll 4
            for (int i = 0; i < HID; ++i) acc += st[b][i] * Wlb[(size_t)(HID + i) * HID + j];
            ft[b][j] = tanhf(acc);
        }
        __syncthreads();
#pragma unroll
        for (int b = 0; b < 2; ++b) {
            float f = blt[j], g = blg[j], hc = blh[j];
#pragma unroll 4
            for (int i = 0; i < HID; ++i) {
                float fv = ft[b][i];
                f  += fv * Wlt[(size_t)i * HID + j];
                g  += fv * Wlg[(size_t)i * HID + j];
                hc += fv * Wlh[(size_t)i * HID + j];
            }
            f = sigf(f); g = tanhf(g); hc = tanhf(hc);
            float tt   = times[b * LSEQ + t];
            float gate = sigf(-f * tt);
            float nw   = gate * g + (1.0f - gate) * hc;
            yhid[(size_t)(b * LSEQ + t) * HID + j] = nw;
            st[b][j] = nw;
        }
        __syncthreads();
    }
}

// ---------------------------------------------------------------------------
// One-time prep: f32 -> bf16 convert, and weight transpose W[K,N]f32 -> Wt[N,K]bf16
// ---------------------------------------------------------------------------
__global__ __launch_bounds__(256)
void f32_to_bf16_kernel(const float* __restrict__ in, unsigned short* __restrict__ out, int n) {
    int i = blockIdx.x * 256 + threadIdx.x;
    if (i < n) out[i] = f2bf(in[i]);
}

__global__ __launch_bounds__(256)
void transpose_bf16_kernel(const float* __restrict__ W, unsigned short* __restrict__ Wt,
                           int K, int N) {
    __shared__ float t[32][33];
    const int col = threadIdx.x;        // 0..31
    const int rq  = threadIdx.y;        // 0..7
    const int n0 = blockIdx.x * 32, k0 = blockIdx.y * 32;
#pragma unroll
    for (int j = 0; j < 32; j += 8)
        t[rq + j][col] = W[(size_t)(k0 + rq + j) * N + n0 + col];
    __syncthreads();
#pragma unroll
    for (int j = 0; j < 32; j += 8)
        Wt[(size_t)(n0 + rq + j) * K + k0 + col] = f2bf(t[col][rq + j]);
}

// ---------------------------------------------------------------------------
extern "C" void kernel_launch(void* const* d_in, const int* in_sizes, int n_in,
                              void* d_out, int out_size, void* d_ws, size_t ws_size,
                              hipStream_t stream) {
    (void)in_sizes; (void)n_in; (void)out_size; (void)ws_size;

    const float* x     = (const float*)d_in[0];
    const float* times = (const float*)d_in[1];
    const float* W_in  = (const float*)d_in[2];
    const float* b_in  = (const float*)d_in[3];
    const float* W_qkv = (const float*)d_in[4];
    const float* b_qkv = (const float*)d_in[5];
    const float* W_ao  = (const float*)d_in[6];
    const float* b_ao  = (const float*)d_in[7];
    const float* W_f1  = (const float*)d_in[8];
    const float* b_f1  = (const float*)d_in[9];
    const float* W_f2  = (const float*)d_in[10];
    const float* b_f2  = (const float*)d_in[11];
    const float* g1    = (const float*)d_in[12];
    const float* be1   = (const float*)d_in[13];
    const float* g2    = (const float*)d_in[14];
    const float* be2   = (const float*)d_in[15];
    const float* W_cp  = (const float*)d_in[16];
    const float* b_cp  = (const float*)d_in[17];
    const float* W_cg  = (const float*)d_in[18];
    const float* b_cg  = (const float*)d_in[19];
    const float* W_lb  = (const float*)d_in[20];
    const float* b_lb  = (const float*)d_in[21];
    const float* W_lt  = (const float*)d_in[22];
    const float* b_lt  = (const float*)d_in[23];
    const float* W_lg  = (const float*)d_in[24];
    const float* b_lg  = (const float*)d_in[25];
    const float* W_lh  = (const float*)d_in[26];
    const float* b_lh  = (const float*)d_in[27];
    const float* W_out = (const float*)d_in[28];
    const float* b_out = (const float*)d_in[29];

    char* base = (char*)d_ws;
    size_t off = 0;
    auto carve = [&](size_t bytes) {
        void* p = base + off;
        off = (off + bytes + 255) & ~(size_t)255;
        return p;
    };
    // f32 buffers
    float* h     = (float*)carve((size_t)MROWS * HID * 4);
    float* ao    = (float*)carve((size_t)MROWS * HID * 4);
    float* ffn   = (float*)carve((size_t)MROWS * HID * 4);
    float* h1    = (float*)carve((size_t)MROWS * HID * 4);
    float* cand  = (float*)carve((size_t)MROWS * HID * 4);
    float* ycg   = (float*)carve((size_t)MROWS * HID * 4);
    float* yhid  = (float*)carve((size_t)MROWS * HID * 4);
    float* h2    = (float*)carve((size_t)MROWS * HID * 4);
    float* alpha = (float*)carve((size_t)MROWS * 4);
    // bf16 activations
    unsigned short* xbf    = (unsigned short*)carve((size_t)MROWS * 512 * 2);
    unsigned short* h_bf   = (unsigned short*)carve((size_t)MROWS * HID * 2);
    unsigned short* qkv_bf = (unsigned short*)carve((size_t)MROWS * 3 * HID * 2);
    unsigned short* aoh_bf = (unsigned short*)carve((size_t)MROWS * HID * 2);
    unsigned short* f1a_bf = (unsigned short*)carve((size_t)MROWS * 4 * HID * 2);
    unsigned short* h1_bf  = (unsigned short*)carve((size_t)MROWS * HID * 2);
    unsigned short* h2_bf  = (unsigned short*)carve((size_t)MROWS * HID * 2);
    // bf16 transposed weights [N,K]
    unsigned short* Wt_in  = (unsigned short*)carve((size_t)512 * HID * 2);
    unsigned short* Wt_qkv = (unsigned short*)carve((size_t)HID * 3 * HID * 2);
    unsigned short* Wt_ao  = (unsigned short*)carve((size_t)HID * HID * 2);
    unsigned short* Wt_f1  = (unsigned short*)carve((size_t)HID * 4 * HID * 2);
    unsigned short* Wt_f2  = (unsigned short*)carve((size_t)4 * HID * HID * 2);
    unsigned short* Wt_cp  = (unsigned short*)carve((size_t)HID * HID * 2);
    unsigned short* Wt_out = (unsigned short*)carve((size_t)HID * 512 * 2);

    float* y = (float*)d_out;   // [B,L,I] f32

    const dim3 blk(256);
    const dim3 tb(32, 8);
    // --- prep: transpose+convert weights, convert x ---
    transpose_bf16_kernel<<<dim3(HID / 32, 512 / 32), tb, 0, stream>>>(W_in, Wt_in, 512, HID);
    transpose_bf16_kernel<<<dim3(3 * HID / 32, HID / 32), tb, 0, stream>>>(W_qkv, Wt_qkv, HID, 3 * HID);
    transpose_bf16_kernel<<<dim3(HID / 32, HID / 32), tb, 0, stream>>>(W_ao, Wt_ao, HID, HID);
    transpose_bf16_kernel<<<dim3(4 * HID / 32, HID / 32), tb, 0, stream>>>(W_f1, Wt_f1, HID, 4 * HID);
    transpose_bf16_kernel<<<dim3(HID / 32, 4 * HID / 32), tb, 0, stream>>>(W_f2, Wt_f2, 4 * HID, HID);
    transpose_bf16_kernel<<<dim3(HID / 32, HID / 32), tb, 0, stream>>>(W_cp, Wt_cp, HID, HID);
    transpose_bf16_kernel<<<dim3(512 / 32, HID / 32), tb, 0, stream>>>(W_out, Wt_out, HID, 512);
    f32_to_bf16_kernel<<<dim3((MROWS * 512) / 256), blk, 0, stream>>>(x, xbf, MROWS * 512);

    // 1) h = x @ W_in + b_in           (f32 + bf16 out)
    gemm_bf16t_kernel<ACT_NONE><<<dim3(HID / 128, MROWS / 128), blk, 0, stream>>>(
        xbf, Wt_in, b_in, h, h_bf, MROWS, HID, 512);
    // 2) qkv = h @ W_qkv + b_qkv       (bf16 only)
    gemm_bf16t_kernel<ACT_NONE><<<dim3(3 * HID / 128, MROWS / 128), blk, 0, stream>>>(
        h_bf, Wt_qkv, b_qkv, nullptr, qkv_bf, MROWS, 3 * HID, HID);
    // 3) attention -> aoh (bf16)
    attn_kernel<<<dim3(2 * NH * (LSEQ / 64)), dim3(128), 0, stream>>>(qkv_bf, aoh_bf);
    // 4) ao = aoh @ W_ao + b_ao        (f32)
    gemm_bf16t_kernel<ACT_NONE><<<dim3(HID / 128, MROWS / 128), blk, 0, stream>>>(
        aoh_bf, Wt_ao, b_ao, ao, nullptr, MROWS, HID, HID);
    // 5) f1a = silu(h @ W_f1 + b_f1)   (bf16)
    gemm_bf16t_kernel<ACT_SILU><<<dim3(4 * HID / 128, MROWS / 128), blk, 0, stream>>>(
        h_bf, Wt_f1, b_f1, nullptr, f1a_bf, MROWS, 4 * HID, HID);
    // 6) ffn = f1a @ W_f2 + b_f2       (f32)
    gemm_bf16t_kernel<ACT_NONE><<<dim3(HID / 128, MROWS / 128), blk, 0, stream>>>(
        f1a_bf, Wt_f2, b_f2, ffn, nullptr, MROWS, HID, 4 * HID);
    // 7) h1 = LN(h + ao + ffn)
    ln_res_kernel<<<dim3(MROWS), blk, 0, stream>>>(h, ao, ffn, g1, be1, h1, h1_bf);
    // 8) alpha = sigmoid(h1 @ W_cg + b_cg)
    cube_gate_gemv<<<dim3(MROWS / 8), blk, 0, stream>>>(h1, W_cg, b_cg, alpha);
    // 9) cand = tanh(h1 @ W_cp + b_cp)
    gemm_bf16t_kernel<ACT_TANH><<<dim3(HID / 128, MROWS / 128), blk, 0, stream>>>(
        h1_bf, Wt_cp, b_cp, cand, nullptr, MROWS, HID, HID);
    // 10) ycg = alpha*cand + (1-alpha)*h1
    cube_blend_kernel<<<dim3((MROWS * HID) / 256), blk, 0, stream>>>(
        alpha, cand, h1, ycg, MROWS * HID);
    // 11) sequential LTC scan -> yhid
    ltc_scan_kernel<<<dim3(1), dim3(1024), 0, stream>>>(
        ycg, times, W_lb, b_lb, W_lt, b_lt, W_lg, b_lg, W_lh, b_lh, yhid);
    // 12) h2 = LN(yhid + ycg)
    ln_res_kernel<<<dim3(MROWS), blk, 0, stream>>>(yhid, ycg, nullptr, g2, be2, h2, h2_bf);
    // 13) y = h2 @ W_out + b_out
    gemm_bf16t_kernel<ACT_NONE><<<dim3(512 / 128, MROWS / 128), blk, 0, stream>>>(
        h2_bf, Wt_out, b_out, y, nullptr, MROWS, 512, HID);
}
